// BFConv2d_69604239999615
// MI455X (gfx1250) — compile-verified, matrix operand order checked by reading, not run on hardware
//
#include <hip/hip_runtime.h>

typedef _Float16 h16;
typedef __attribute__((ext_vector_type(8)))  _Float16 v8h;
typedef __attribute__((ext_vector_type(16))) _Float16 v16h;
typedef __attribute__((ext_vector_type(8)))  float    v8f;

#define C_IN   64
#define K_OUT  64
#define WIMG   112
#define HW     12544          // 112*112
#define NIMG   32
#define XN     25690112       // 32*64*112*112
#define XG     713615         // ceil(XN/36)
#define WPAD   114
#define PLANE  12996          // 114*114 padded spatial plane
#define TILE_N 128
#define LPAD   48             // LDS row stride in halfs (96B, 16B-aligned)

// ---------------- BFP helpers ----------------
// scale = 2^(floor(log2(m)) - 7); exact via exponent field for normal m.
__device__ __forceinline__ float bfp_scale(float m) {
    unsigned eb = (m > 0.0f) ? ((__float_as_uint(m) >> 23) & 0xFFu) : 127u;
    return __uint_as_float((eb - 7u) << 23);
}
// q = round_half_even(v / s) * s ; 1/s of a power-of-two via exponent flip.
__device__ __forceinline__ h16 bfp_apply(float v, float s) {
    float r = __uint_as_float((254u << 23) - __float_as_uint(s));
    return (h16)(__builtin_rintf(v * r) * s);
}

// ---------------- CDNA5 async copy to LDS ----------------
__device__ __forceinline__ void async_b128(unsigned lds_off, const void* gptr) {
    asm volatile("global_load_async_to_lds_b128 %0, %1, off"
                 :: "v"(lds_off), "v"((unsigned long long)(uintptr_t)gptr)
                 : "memory");
}
__device__ __forceinline__ void wait_async0() {
    asm volatile("s_wait_asynccnt 0x0" ::: "memory");
}
// low 32 bits of a flat shared pointer == LDS byte offset (ISA 10.2 aperture map)
__device__ __forceinline__ unsigned lds_off(const void* p) {
    return (unsigned)(uintptr_t)p;
}

// ---- kernel 1: quantize weights into [rs][k][c] f16 layout -----------------
__global__ void wquant_kernel(const float* __restrict__ w, h16* __restrict__ wq) {
    int g = blockIdx.x * blockDim.x + threadIdx.x;
    if (g >= 1024) return;                    // 64*64*9 / 36 = 1024 exact
    int base = g * 36;
    float vals[36];
    float m = 0.0f;
#pragma unroll
    for (int i = 0; i < 36; ++i) { vals[i] = w[base + i]; m = fmaxf(m, fabsf(vals[i])); }
    float s = bfp_scale(m);
#pragma unroll
    for (int i = 0; i < 36; ++i) {
        int flat = base + i;                  // flat = k*576 + c*9 + rs
        int k   = flat / 576;
        int rem = flat - k * 576;
        int c   = rem / 9;
        int rs  = rem - c * 9;
        wq[rs * 4096 + k * 64 + c] = bfp_apply(vals[i], s);
    }
}

// ---- kernel 2: per-group scales for x --------------------------------------
__global__ void xscale_kernel(const float* __restrict__ x, float* __restrict__ sc) {
    int g = blockIdx.x * blockDim.x + threadIdx.x;
    if (g >= XG) return;
    int base = g * 36;
    float m = 0.0f;
#pragma unroll
    for (int i = 0; i < 36; ++i) {
        int idx = base + i;
        if (idx < XN) m = fmaxf(m, fabsf(x[idx]));
    }
    sc[g] = bfp_scale(m);
}

// ---- kernel 3: zero-fill padded xq (borders stay zero) ---------------------
__global__ void xzero_kernel(h16* __restrict__ xq) {
    int i = blockIdx.x * blockDim.x + threadIdx.x;   // one int4 (8 halfs) each
    ((int4*)xq)[i] = make_int4(0, 0, 0, 0);          // grid sized exactly
}

// ---- kernel 4: quantize x NCHW f32 -> padded NHWC f16 ----------------------
// xq[n][(h+1)*114 + (w+1)][c], c innermost (64 halfs = 128B rows)
__global__ __launch_bounds__(256) void xquant_kernel(
        const float* __restrict__ x, const float* __restrict__ xs,
        h16* __restrict__ xq) {
    int bx = blockIdx.x;                 // 32 * 49 blocks
    int n  = bx / 49;
    int sp = (bx - n * 49) * 256 + threadIdx.x;      // 0..12543
    int h  = sp / WIMG;
    int w  = sp - h * WIMG;
    unsigned pk[32];
#pragma unroll
    for (int cp = 0; cp < 32; ++cp) {
        int c    = cp * 2;
        int idx0 = (n * C_IN + c) * HW + sp;
        int idx1 = idx0 + HW;
        h16 q0 = bfp_apply(x[idx0], xs[(unsigned)idx0 / 36u]);
        h16 q1 = bfp_apply(x[idx1], xs[(unsigned)idx1 / 36u]);
        union { h16 h[2]; unsigned u; } u2;
        u2.h[0] = q0; u2.h[1] = q1;
        pk[cp] = u2.u;
    }
    size_t dst = ((size_t)n * PLANE + (h + 1) * WPAD + (w + 1)) * 64;
    int4* dp = (int4*)(xq + dst);        // 128B-aligned
#pragma unroll
    for (int i = 0; i < 8; ++i)
        dp[i] = make_int4((int)pk[4*i], (int)pk[4*i+1], (int)pk[4*i+2], (int)pk[4*i+3]);
}

// ---- kernel 5: implicit-GEMM conv with WMMA + async LDS fills --------------
__global__ __launch_bounds__(256) void conv_wmma_kernel(
        const h16* __restrict__ xq, const h16* __restrict__ wq,
        const float* __restrict__ bias, float* __restrict__ out) {
    __shared__ __align__(16) h16 At[64][LPAD];      // A: 64(k) x 32(c)
    __shared__ __align__(16) h16 Bt[TILE_N][LPAD];  // B: 128(n) x 32(c)
    __shared__ float Bs[K_OUT];

    int tid = threadIdx.x;
    if (tid < K_OUT) Bs[tid] = bias[tid];

    int p0   = blockIdx.x * TILE_N;      // 12544 % 128 == 0: no image crossing
    int nimg = p0 / HW;
    int hw0  = p0 - nimg * HW;

    int lane = tid & 31;
    int wave = tid >> 5;

    // B-fill roles: 512 16B-chunks; thread t -> chunks t and t+256
    int colc = (tid & 3) * 8;            // half offset within row
    int j0   = tid >> 2;                 // rows j0 and j0+64
    int hwa  = hw0 + j0;
    int ha   = hwa / WIMG, wa = hwa - ha * WIMG;
    int spA  = (ha + 1) * WPAD + (wa + 1);          // padded center offset
    int hwb  = hwa + 64;
    int hb   = hwb / WIMG, wb = hwb - hb * WIMG;
    int spB  = (hb + 1) * WPAD + (wb + 1);
    const h16* xqn = xq + (size_t)nimg * PLANE * 64;

    unsigned ldsA = lds_off(&At[tid >> 2][colc]);   // threads 0..127 use this
    unsigned ldsB0 = lds_off(&Bt[j0][colc]);
    unsigned ldsB1 = lds_off(&Bt[j0 + 64][colc]);

    v8f acc[4] = {};

    for (int rs = 0; rs < 9; ++rs) {
        int dr = rs / 3 - 1;
        int dc = rs - (rs / 3) * 3 - 1;
        int rofs = dr * WPAD + dc;                   // padded-plane delta
        const h16* wsrc = wq + rs * 4096;
        for (int cc = 0; cc < 2; ++cc) {
            int cbase = cc * 32;
            __syncthreads();                         // prev LDS reads done
            if (tid < 128)                           // A tile: 128 chunks
                async_b128(ldsA, wsrc + (tid >> 2) * 64 + cbase + colc);
            async_b128(ldsB0, xqn + (size_t)(spA + rofs) * 64 + cbase + colc);
            async_b128(ldsB1, xqn + (size_t)(spB + rofs) * 64 + cbase + colc);
            wait_async0();
            __syncthreads();

            // WMMA fragments per ISA 16-bit layout:
            // lane<16 -> K chunks {0..7,16..23}; lane>=16 -> {8..15,24..31}
            int kb = (lane < 16) ? 0 : 8;
            int jn = wave * 16 + (lane & 15);
            v8h blo = *(const v8h*)&Bt[jn][kb];
            v8h bhi = *(const v8h*)&Bt[jn][kb + 16];
            v16h bfrag = __builtin_shufflevector(blo, bhi,
                0,1,2,3,4,5,6,7,8,9,10,11,12,13,14,15);
#pragma unroll
            for (int mt = 0; mt < 4; ++mt) {
                int m = mt * 16 + (lane & 15);
                v8h alo = *(const v8h*)&At[m][kb];
                v8h ahi = *(const v8h*)&At[m][kb + 16];
                v16h afrag = __builtin_shufflevector(alo, ahi,
                    0,1,2,3,4,5,6,7,8,9,10,11,12,13,14,15);
                acc[mt] = __builtin_amdgcn_wmma_f32_16x16x32_f16(
                    false, afrag, false, bfrag, (short)0, acc[mt], false, false);
            }
        }
    }

    // epilogue: C/D layout — lane<16 rows v, lane>=16 rows v+8; col = lane%16
    int jn    = wave * 16 + (lane & 15);
    int hwout = hw0 + jn;
    int mofs  = (lane < 16) ? 0 : 8;
#pragma unroll
    for (int mt = 0; mt < 4; ++mt) {
#pragma unroll
        for (int v = 0; v < 8; ++v) {
            int k = mt * 16 + mofs + v;
            __builtin_nonte\
mporal_store(acc[mt][v] + Bs[k], &out[(nimg * C_IN + k) * HW + hwout]);
        }
    }
}

extern "C" void kernel_launch(void* const* d_in, const int* in_sizes, int n_in,
                              void* d_out, int out_size, void* d_ws, size_t ws_size,
                              hipStream_t stream) {
    const float* x    = (const float*)d_in[0];
    const float* w    = (const float*)d_in[1];
    const float* bias = (const float*)d_in[2];
    float* out = (float*)d_out;

    // workspace: [wq f16 72KB][x scales f32 2.9MB][padded NHWC xq f16 53MB]
    h16*   wq = (h16*)d_ws;
    float* xs = (float*)((char*)d_ws + 73728);
    size_t xq_off = (73728 + (size_t)XG * 4 + 255) & ~(size_t)255;
    h16*   xq = (h16*)((char*)d_ws + xq_off);

    wquant_kernel<<<4, 256, 0, stream>>>(w, wq);
    xscale_kernel<<<(XG + 255) / 256, 256, 0, stream>>>(x, xs);
    // padded xq: 32*12996*64 halfs = 3,326,976 int4s = 12,996 blocks x 256
    xzero_kernel<<<12996, 256, 0, stream>>>(xq);
    xquant_kernel<<<NIMG * 49, 256, 0, stream>>>(x, xs, xq);
    conv_wmma_kernel<<<(NIMG * HW) / TILE_N, 256, 0, stream>>>(xq, wq, bias, out);
}
